// RQVAEModule_55963423867261
// MI455X (gfx1250) — matrix operand order repaired and checked by reading, not compiled, not run on hardware
//
#include <hip/hip_runtime.h>
#include <hip/hip_bf16.h>

// ---------------------------------------------------------------------------
// RQ-VAE forward on gfx1250 (MI455X).
// All 8 MLP GEMMs run on v_wmma_f32_16x16x32_f16 (f16 A/B, f32 accumulate)
// with fused bias + ReLU. A tiles are staged into LDS with the CDNA5 async
// global->LDS path (ASYNCcnt) and double-buffered; weights are L2-resident
// and read directly. Activations kept f16 to respect the 23.3 TB/s roofline.
// Residual VQ is a wave-per-row VALU kernel (codebooks are L2-resident).
// ---------------------------------------------------------------------------

typedef __attribute__((ext_vector_type(16))) _Float16 v16h;
typedef __attribute__((ext_vector_type(8)))  _Float16 v8h;
typedef __attribute__((ext_vector_type(8)))  float    v8f;

#define RQ_B       65536
#define RQ_EDIM    64
#define RQ_STAGES  4
#define RQ_CODES   256
#define RQ_BETA    0.25f

// ---------------------------------------------------------------------------
// CDNA5 async global->LDS copy (GLOBAL_LOAD_ASYNC_TO_LDS_B128, ASYNCcnt).
// Builtin prototype (from clang diagnostic): (v4i AS1*, v4i AS3*, Ii, Ii).
// Guarded: falls back to synchronous staging if the builtin is unavailable.
// ---------------------------------------------------------------------------
#if defined(__has_builtin)
#if __has_builtin(__builtin_amdgcn_global_load_async_to_lds_b128)
#define RQ_HAS_ASYNC 1
#endif
#endif
#ifndef RQ_HAS_ASYNC
#define RQ_HAS_ASYNC 0
#endif

typedef int rq_v4i __attribute__((vector_size(16)));
typedef rq_v4i __attribute__((address_space(1)))* rq_gp128;
typedef rq_v4i __attribute__((address_space(3)))* rq_lp128;

__device__ __forceinline__ void rq_cp16_g2l(const _Float16* g, _Float16* l) {
#if RQ_HAS_ASYNC
  // Global AS pointer: flat value == global value, so inttoptr is exact
  // (also drops const). LDS pointer: addrspacecast strips the aperture.
  rq_gp128 gp = (rq_gp128)(uintptr_t)g;
  rq_lp128 lp = (rq_lp128)l;
  __builtin_amdgcn_global_load_async_to_lds_b128(gp, lp, 0, 0);
#else
  *(v8h*)l = *(const v8h*)g;
#endif
}

__device__ __forceinline__ void rq_wait_async() {
#if RQ_HAS_ASYNC
#if __has_builtin(__builtin_amdgcn_s_wait_asynccnt)
  __builtin_amdgcn_s_wait_asynccnt(0);
#else
  asm volatile("s_wait_asynccnt 0" ::: "memory");
#endif
#endif
}

// ---------------------------------------------------------------------------
// f32 -> f16 conversion (x and weights)
// ---------------------------------------------------------------------------
__global__ void rqvae_f32_to_f16(const float* __restrict__ src,
                                 _Float16* __restrict__ dst, int n) {
  int i = blockIdx.x * blockDim.x + threadIdx.x;
  if (i < n) dst[i] = (_Float16)src[i];
}

// ---------------------------------------------------------------------------
// WMMA fragment loaders (wave32; contiguous k = 16h+s for both A and B, not the ISA table order)
//   A (16x32 f16, MxK): lanes 0-15 row M=lane  hold K{0..15}
//                       lanes16-31 row M=l-16  hold K{16..31}
//   B (32x16 f16, KxN): lanes 0-15 col N=lane  hold K{0..15}
//                       lanes16-31 col N=l-16  hold K{16..31}
//   W is row-major (N x K) = exactly column-major B.
// ---------------------------------------------------------------------------
__device__ __forceinline__ v16h rq_frag_a_lds(const _Float16* __restrict__ base,
                                              int stride, int r0, int k32, int lane) {
  int r    = r0 + (lane & 15);
  int ksel = (lane >> 4) * 16;
  const _Float16* p = base + r * stride + k32 + ksel;
  v8h lo = *(const v8h*)p;          // K = k32+ksel .. +7
  v8h hi = *(const v8h*)(p + 8);    // K = k32+8+ksel .. +7
  v16h out;
#pragma unroll
  for (int i = 0; i < 8; ++i) { out[i] = lo[i]; out[8 + i] = hi[i]; }
  return out;
}

__device__ __forceinline__ v16h rq_load_b(const _Float16* __restrict__ W,
                                          int ldw, int n0, int k0, int lane) {
  int n  = n0 + (lane & 15);
  int ks = k0 + (lane >> 4) * 16;
  return *(const v16h*)(W + (size_t)n * ldw + ks);   // 32B, aligned
}

// ---------------------------------------------------------------------------
// Fused GEMM: out[M,N] = act( A[M,K] * W[N,K]^T + bias[N] )
// 256 threads = 8 waves (4 M x 2 N). Wave tile 32 x (16*FN); block tile
// 128 x (32*FN). A staged in LDS (async, double-buffered, 64-K chunks);
// W read direct from global (L2-resident). M%128==0, N%(32*FN)==0, K%64==0.
// ---------------------------------------------------------------------------
template <bool RELU, bool OUTF32, int FN>
__global__ void __launch_bounds__(256)
rqvae_wmma_gemm(const _Float16* __restrict__ A,
                const _Float16* __restrict__ W,
                const float* __restrict__ bias,
                _Float16* __restrict__ outH,
                float* __restrict__ outF,
                int M, int N, int K) {
  constexpr int KC = 64;   // staged K chunk
  constexpr int LP = 8;    // pad: row stride 72 halves = 36 dwords -> no bank conflicts
  __shared__ _Float16 lA[2][128][KC + LP];

  const int lane  = threadIdx.x & 31;
  const int wave  = threadIdx.x >> 5;
  const int waveM = wave & 3;   // 0..3
  const int waveN = wave >> 2;  // 0..1
  const int rowBlock = blockIdx.x * 128;
  const int colBase  = blockIdx.y * (32 * FN) + waveN * (16 * FN);

  v8f acc[2][FN];
#pragma unroll
  for (int i = 0; i < 2; ++i)
#pragma unroll
    for (int j = 0; j < FN; ++j)
#pragma unroll
      for (int e = 0; e < 8; ++e) acc[i][j][e] = 0.0f;

  const int nCh = K / KC;

  auto loadA = [&](int c, int b) {
    const _Float16* src = A + (size_t)rowBlock * K + c * KC;
#pragma unroll
    for (int it = 0; it < 4; ++it) {           // 128 rows x 64 halves, 16B/lane
      int idx = (int)threadIdx.x + it * 256;   // 0..1023
      int row = idx >> 3;
      int col = (idx & 7) * 8;
      rq_cp16_g2l(src + (size_t)row * K + col, &lA[b][row][col]);
    }
  };

  loadA(0, 0);
  rq_wait_async();
  __syncthreads();

  for (int c = 0; c < nCh; ++c) {
    const int b = c & 1;
    if (c + 1 < nCh) loadA(c + 1, b ^ 1);      // async into other buffer
#pragma unroll
    for (int s = 0; s < 2; ++s) {
      const int k32 = s * 32;
      v16h a0 = rq_frag_a_lds(&lA[b][0][0], KC + LP, waveM * 32,      k32, lane);
      v16h a1 = rq_frag_a_lds(&lA[b][0][0], KC + LP, waveM * 32 + 16, k32, lane);
#pragma unroll
      for (int fj = 0; fj < FN; ++fj) {
        v16h bb = rq_load_b(W, K, colBase + fj * 16, c * KC + k32, lane);
        acc[0][fj] = __builtin_amdgcn_wmma_f32_16x16x32_f16(false, a0, false, bb,
                                                            (short)0, acc[0][fj], false, false);
        acc[1][fj] = __builtin_amdgcn_wmma_f32_16x16x32_f16(false, a1, false, bb,
                                                            (short)0, acc[1][fj], false, false);
      }
    }
    rq_wait_async();
    __syncthreads();
  }

  // D layout: VGPR i -> lanes0-15 (M=i, N=lane), lanes16-31 (M=8+i, N=lane-16)
  const int mOff = (lane >> 4) * 8;
  const int nLo  = lane & 15;
#pragma unroll
  for (int fi = 0; fi < 2; ++fi) {
#pragma unroll
    for (int fj = 0; fj < FN; ++fj) {
      int n = colBase + fj * 16 + nLo;
      float bv = bias[n];
#pragma unroll
      for (int i = 0; i < 8; ++i) {
        int m = rowBlock + waveM * 32 + fi * 16 + mOff + i;
        float v = acc[fi][fj][i] + bv;
        if (RELU) v = fmaxf(v, 0.0f);
        if (OUTF32) outF[(size_t)m * N + n] = v;
        else        outH[(size_t)m * N + n] = (_Float16)v;
      }
    }
  }
}

// ---------------------------------------------------------------------------
// Residual VQ: one wave per row. Residual (64 f32) lives in registers in
// every lane; each lane scores 8 codes (d = |c|^2 - 2 r.c ; |r|^2 is
// argmin-invariant); butterfly shfl_xor argmin; loss via LDS + 1 atomic/block.
// ---------------------------------------------------------------------------
__global__ void __launch_bounds__(256)
rqvae_quantize(const _Float16* __restrict__ z,
               const float* __restrict__ codebooks,
               _Float16* __restrict__ xq,
               float* __restrict__ out_tail,   // d_out + B*768 : [loss, idx...]
               float* __restrict__ loss_acc) {
  const int lane = threadIdx.x & 31;
  const int wave = threadIdx.x >> 5;
  const int row  = blockIdx.x * 8 + wave;
  __shared__ float wloss[8];

  float r[RQ_EDIM];
  const _Float16* zrow = z + (size_t)row * RQ_EDIM;
#pragma unroll
  for (int j = 0; j < RQ_EDIM; ++j) r[j] = (float)zrow[j];

  float lsum = 0.0f;
  for (int s = 0; s < RQ_STAGES; ++s) {
    const float* cb = codebooks + (size_t)s * RQ_CODES * RQ_EDIM;
    float best = 3.4e38f;
    int bestIdx = 0;
    for (int c = 0; c < RQ_CODES / 32; ++c) {
      const int code = lane * (RQ_CODES / 32) + c;
      const float* cp = cb + (size_t)code * RQ_EDIM;
      float dot = 0.0f, nrm = 0.0f;
#pragma unroll
      for (int j = 0; j < RQ_EDIM; ++j) {
        float cv = cp[j];
        dot = fmaf(r[j], cv, dot);
        nrm = fmaf(cv, cv, nrm);
      }
      float d = nrm - 2.0f * dot;
      if (d < best) { best = d; bestIdx = code; }
    }
    for (int off = 16; off > 0; off >>= 1) {   // wave32 butterfly argmin
      float ob = __shfl_xor(best, off, 32);
      int   oi = __shfl_xor(bestIdx, off, 32);
      if (ob < best || (ob == best && oi < bestIdx)) { best = ob; bestIdx = oi; }
    }
    const float* qp = cb + (size_t)bestIdx * RQ_EDIM;
    float sq = 0.0f;
#pragma unroll
    for (int j = 0; j < RQ_EDIM; ++j) {
      float qv = qp[j];
      float df = qv - r[j];
      sq = fmaf(df, df, sq);
      r[j] -= qv;
    }
    lsum += sq;
    if (lane == 0) out_tail[1 + (size_t)row * RQ_STAGES + s] = (float)bestIdx;
  }

  _Float16* xqrow = xq + (size_t)row * RQ_EDIM;     // x_q = z - final residual
#pragma unroll
  for (int t = 0; t < 2; ++t) {
    int j = lane * 2 + t;
    xqrow[j] = (_Float16)((float)zrow[j] - r[j]);
  }

  if (lane == 0) wloss[wave] = lsum;
  __syncthreads();
  if (threadIdx.x == 0) {
    float t = 0.0f;
#pragma unroll
    for (int w = 0; w < 8; ++w) t += wloss[w];
    atomicAdd(loss_acc, t);
  }
}

__global__ void rqvae_finalize_loss(const float* __restrict__ acc,
                                    float* __restrict__ out, float scale) {
  if (threadIdx.x == 0 && blockIdx.x == 0) out[0] = acc[0] * scale;
}

// ---------------------------------------------------------------------------
// Host orchestration
// ---------------------------------------------------------------------------
static inline void launch_gemm(bool relu, bool outf32,
                               const _Float16* A, const _Float16* W,
                               const float* bias, _Float16* outH, float* outF,
                               int M, int N, int K, hipStream_t s) {
  if (N % 128 == 0) {
    dim3 grid(M / 128, N / 128);
    if (relu)
      rqvae_wmma_gemm<true,  false, 4><<<grid, 256, 0, s>>>(A, W, bias, outH, outF, M, N, K);
    else if (outf32)
      rqvae_wmma_gemm<false, true,  4><<<grid, 256, 0, s>>>(A, W, bias, outH, outF, M, N, K);
    else
      rqvae_wmma_gemm<false, false, 4><<<grid, 256, 0, s>>>(A, W, bias, outH, outF, M, N, K);
  } else {
    dim3 grid(M / 128, N / 64);
    if (relu)
      rqvae_wmma_gemm<true,  false, 2><<<grid, 256, 0, s>>>(A, W, bias, outH, outF, M, N, K);
    else if (outf32)
      rqvae_wmma_gemm<false, true,  2><<<grid, 256, 0, s>>>(A, W, bias, outH, outF, M, N, K);
    else
      rqvae_wmma_gemm<false, false, 2><<<grid, 256, 0, s>>>(A, W, bias, outH, outF, M, N, K);
  }
}

extern "C" void kernel_launch(void* const* d_in, const int* in_sizes, int n_in,
                              void* d_out, int out_size, void* d_ws, size_t ws_size,
                              hipStream_t stream) {
  (void)in_sizes; (void)n_in; (void)out_size; (void)ws_size;

  const float* x   = (const float*)d_in[0];
  const float* ew[4] = { (const float*)d_in[1], (const float*)d_in[3],
                         (const float*)d_in[5], (const float*)d_in[7] };
  const float* eb[4] = { (const float*)d_in[2], (const float*)d_in[4],
                         (const float*)d_in[6], (const float*)d_in[8] };
  const float* dw[4] = { (const float*)d_in[9],  (const float*)d_in[11],
                         (const float*)d_in[13], (const float*)d_in[15] };
  const float* db[4] = { (const float*)d_in[10], (const float*)d_in[12],
                         (const float*)d_in[14], (const float*)d_in[16] };
  const float* codebooks = (const float*)d_in[17];

  const int encDims[5] = {768, 512, 256, 128, 64};
  const int decDims[5] = {64, 128, 256, 512, 768};

  // Workspace layout (halves unless noted)
  const size_t actElems = (size_t)RQ_B * 768;
  _Float16* bufA = (_Float16*)d_ws;                 // B*768 halves
  _Float16* bufB = bufA + actElems;                 // B*768 halves
  _Float16* wgt  = bufB + actElems;                 // 1,130,496 halves total
  size_t woff = 0;
  _Float16* ewh[4]; _Float16* dwh[4];
  for (int i = 0; i < 4; ++i) {
    ewh[i] = wgt + woff; woff += (size_t)encDims[i + 1] * encDims[i];
  }
  for (int i = 0; i < 4; ++i) {
    dwh[i] = wgt + woff; woff += (size_t)decDims[i + 1] * decDims[i];
  }
  float* loss_acc = (float*)(wgt + woff);

  float* outF    = (float*)d_out;                   // [B,768]
  float* outTail = outF + actElems;                 // [loss, indices(B*4)]

  (void)hipMemsetAsync(loss_acc, 0, sizeof(float), stream);

  // Convert x and weights to f16
  {
    int n = (int)actElems;
    rqvae_f32_to_f16<<<(n + 255) / 256, 256, 0, stream>>>(x, bufA, n);
  }
  for (int i = 0; i < 4; ++i) {
    int n = encDims[i + 1] * encDims[i];
    rqvae_f32_to_f16<<<(n + 255) / 256, 256, 0, stream>>>(ew[i], ewh[i], n);
    int m = decDims[i + 1] * decDims[i];
    rqvae_f32_to_f16<<<(m + 255) / 256, 256, 0, stream>>>(dw[i], dwh[i], m);
  }

  // Encoder: bufA(768) -> bufB(512) -> bufA(256) -> bufB(128) -> bufA(64)=z
  _Float16* cur = bufA; _Float16* nxt = bufB;
  for (int i = 0; i < 4; ++i) {
    bool relu = (i < 3);
    launch_gemm(relu, false, cur, ewh[i], eb[i], nxt, nullptr,
                RQ_B, encDims[i + 1], encDims[i], stream);
    _Float16* t = cur; cur = nxt; nxt = t;
  }
  _Float16* zbuf  = cur;   // B x 64 (f16)
  _Float16* xqbuf = nxt;

  // Residual quantization (writes x_q f16, indices + loss pieces)
  rqvae_quantize<<<RQ_B / 8, 256, 0, stream>>>(zbuf, codebooks, xqbuf,
                                               outTail, loss_acc);

  // Decoder: xq(64) -> 128 -> 256 -> 512 -> d_out(768, f32)
  cur = xqbuf; nxt = zbuf;
  for (int i = 0; i < 3; ++i) {
    launch_gemm(true, false, cur, dwh[i], db[i], nxt, nullptr,
                RQ_B, decDims[i + 1], decDims[i], stream);
    _Float16* t = cur; cur = nxt; nxt = t;
  }
  launch_gemm(false, true, cur, dwh[3], db[3], nullptr, outF,
              RQ_B, decDims[4], decDims[3], stream);

  // rq_loss = mean_s[(1+beta) * mean_{B*64}(sq_s)]
  const float scale = (1.0f + RQ_BETA) /
                      ((float)RQ_STAGES * (float)RQ_B * (float)RQ_EDIM);
  rqvae_finalize_loss<<<1, 32, 0, stream>>>(loss_acc, outF + actElems, scale);
}
